// Net_45629732553070
// MI455X (gfx1250) — matrix operand order repaired and checked by reading, not compiled, not run on hardware
//
#include <hip/hip_runtime.h>
#include <hip/hip_fp16.h>

typedef _Float16 h16;
typedef __attribute__((ext_vector_type(8)))  _Float16 v8h;
typedef __attribute__((ext_vector_type(16))) _Float16 v16h;
typedef __attribute__((ext_vector_type(8)))  float    v8f;

#define EPS_NORM 1e-12f
#define BN_EPS   1e-5f

// ---------------- row L2-normalize + convert to f16 (with zero row/col pad) ----------------
__global__ __launch_bounds__(256) void norm_rows_f16(
    const float* __restrict__ in, h16* __restrict__ out,
    int Rvalid, int D, int Dpad)
{
  const int r = blockIdx.x;
  const int t = threadIdx.x;
  h16* orow = out + (size_t)r * Dpad;
  if (r >= Rvalid) {                      // zero-pad rows beyond the real matrix
    for (int c = t; c < Dpad; c += 256) orow[c] = (h16)0.f;
    return;
  }
  const float* irow = in + (size_t)r * D;
  float ss = 0.f;
  for (int c = t; c < D; c += 256) { float v = irow[c]; ss += v * v; }
  __shared__ float red[256];
  red[t] = ss; __syncthreads();
  for (int s = 128; s > 0; s >>= 1) { if (t < s) red[t] += red[t + s]; __syncthreads(); }
  const float scale = 1.f / (sqrtf(red[0]) + EPS_NORM);
  for (int c = t; c < D; c += 256) orow[c] = (h16)(irow[c] * scale);
  for (int c = D + t; c < Dpad; c += 256) orow[c] = (h16)0.f;
}

// ---------------- training-mode BatchNorm statistics over the batch dim ----------------
__global__ __launch_bounds__(256) void bn_stats(
    const float* __restrict__ h, int Bn, int C,
    float* __restrict__ mean, float* __restrict__ istd)
{
  const int c = blockIdx.x;
  const int t = threadIdx.x;
  float s = 0.f, sq = 0.f;
  for (int b = t; b < Bn; b += 256) {
    float v = h[(size_t)b * C + c];
    s += v; sq += v * v;
  }
  __shared__ float rs[256], rq[256];
  rs[t] = s; rq[t] = sq; __syncthreads();
  for (int st = 128; st > 0; st >>= 1) {
    if (t < st) { rs[t] += rs[t + st]; rq[t] += rq[t + st]; }
    __syncthreads();
  }
  if (t == 0) {
    float mu  = rs[0] / (float)Bn;
    float var = rq[0] / (float)Bn - mu * mu;    // population var, matches jnp.var
    mean[c] = mu;
    istd[c] = rsqrtf(var + BN_EPS);
  }
}

// ---------------- fused: apply BN, then L2-normalize the row, emit f16 (padded) ----------------
__global__ __launch_bounds__(128) void bn_norm_rows_f16(
    const float* __restrict__ h, const float* __restrict__ mean,
    const float* __restrict__ istd, const float* __restrict__ gamma,
    const float* __restrict__ beta, h16* __restrict__ out,
    int C, int Cpad)
{
  const int r = blockIdx.x;
  const int t = threadIdx.x;
  const float* irow = h + (size_t)r * C;
  float y[4];
  int nc = 0;
  float ss = 0.f;
  for (int c = t; c < C; c += 128) {
    float v = (irow[c] - mean[c]) * istd[c] * gamma[c] + beta[c];
    y[nc++] = v; ss += v * v;
  }
  __shared__ float red[128];
  red[t] = ss; __syncthreads();
  for (int s = 64; s > 0; s >>= 1) { if (t < s) red[t] += red[t + s]; __syncthreads(); }
  const float scale = 1.f / (sqrtf(red[0]) + EPS_NORM);
  h16* orow = out + (size_t)r * Cpad;
  nc = 0;
  for (int c = t; c < C; c += 128) orow[c] = (h16)(y[nc++] * scale);
  for (int c = C + t; c < Cpad; c += 128) orow[c] = (h16)0.f;
}

// ---------------- WMMA GEMM: C[M,Nvalid] = A[M,K] * Bm[Npad,K]^T, optional clamp [0,1] --------
// 10 waves per block; each wave owns 2 N-tiles (n, n+160) and 2 M-tiles -> 4 v8f accumulators.
// K must be a multiple of 32; M a multiple of 32; Npad fixed at 320.
__global__ __launch_bounds__(320) void wmma_gemm_cos(
    const h16* __restrict__ A,    // [M, K] row-major (unit rows)
    const h16* __restrict__ Bm,   // [320, K] row-major = B^T (unit rows, zero-padded)
    float* __restrict__ C,        // [M, ldc]
    int K, int Nvalid, int ldc, int clampOut)
{
  const int wave = (int)(threadIdx.x >> 5);   // 0..9
  const int lane = (int)(threadIdx.x & 31);
  const int m0   = blockIdx.x * 32;           // two 16-row tiles per block
  const int nA   = wave * 16;                 // first N tile
  const int nB   = (wave + 10) * 16;          // second N tile

  const int mRow   = lane & 15;
  const int hiHalf = lane >> 4;
  const int kA = hiHalf << 4;   // A frag: lane-group K offset (0 / 16)
  const int kB = hiHalf << 4;   // B frag: lane-group K offset (0 / 16)

  const h16* a0p = A  + (size_t)(m0 + mRow) * K;
  const h16* a1p = A  + (size_t)(m0 + 16 + mRow) * K;
  const h16* b0p = Bm + (size_t)(nA + mRow) * K + kB;
  const h16* b1p = Bm + (size_t)(nB + mRow) * K + kB;

  v8f acc00 = {}, acc01 = {}, acc10 = {}, acc11 = {};

  for (int k = 0; k < K; k += 32) {
    // A 16x32 f16 fragment: halves 0..7 = K[kA..kA+7], halves 8..15 = K[kA+8..kA+15]
    v8h a0lo = *(const v8h*)(a0p + k + kA);
    v8h a0hi = *(const v8h*)(a0p + k + kA + 8);
    v8h a1lo = *(const v8h*)(a1p + k + kA);
    v8h a1hi = *(const v8h*)(a1p + k + kA + 8);
    v16h fa0 = __builtin_shufflevector(a0lo, a0hi, 0,1,2,3,4,5,6,7,8,9,10,11,12,13,14,15);
    v16h fa1 = __builtin_shufflevector(a1lo, a1hi, 0,1,2,3,4,5,6,7,8,9,10,11,12,13,14,15);
    // B 32x16 f16 fragment: lane = column, 16 contiguous halves of K (kB already applied)
    v16h fb0 = *(const v16h*)(b0p + k);
    v16h fb1 = *(const v16h*)(b1p + k);

    acc00 = __builtin_amdgcn_wmma_f32_16x16x32_f16(false, fa0, false, fb0, (short)0, acc00, false, false);
    acc01 = __builtin_amdgcn_wmma_f32_16x16x32_f16(false, fa0, false, fb1, (short)0, acc01, false, false);
    acc10 = __builtin_amdgcn_wmma_f32_16x16x32_f16(false, fa1, false, fb0, (short)0, acc10, false, false);
    acc11 = __builtin_amdgcn_wmma_f32_16x16x32_f16(false, fa1, false, fb1, (short)0, acc11, false, false);
  }

  // C/D layout: VGPR i -> M = i + 8*(lane>=16), N = lane&15
  const int nOff = lane & 15;
  const int mSub = hiHalf * 8;
  const int colA = nA + nOff, colB = nB + nOff;
  #pragma unroll
  for (int i = 0; i < 8; ++i) {
    int mA = m0 + mSub + i;
    int mB = m0 + 16 + mSub + i;
    float v00 = acc00[i], v01 = acc01[i], v10 = acc10[i], v11 = acc11[i];
    if (clampOut) {
      v00 = fminf(fmaxf(v00, 0.f), 1.f);
      v01 = fminf(fmaxf(v01, 0.f), 1.f);
      v10 = fminf(fmaxf(v10, 0.f), 1.f);
      v11 = fminf(fmaxf(v11, 0.f), 1.f);
    }
    if (colA < Nvalid) { C[(size_t)mA * ldc + colA] = v00; C[(size_t)mB * ldc + colA] = v10; }
    if (colB < Nvalid) { C[(size_t)mA * ldc + colB] = v01; C[(size_t)mB * ldc + colB] = v11; }
  }
}

// ---------------- fused BN2 + fc3 (plain dot product, tiny) ----------------
__global__ __launch_bounds__(32) void bn_fc3(
    const float* __restrict__ h2, const float* __restrict__ mean,
    const float* __restrict__ istd, const float* __restrict__ gamma,
    const float* __restrict__ beta, const float* __restrict__ w3,
    float* __restrict__ out, int C)
{
  const int b = blockIdx.x;
  const int lane = threadIdx.x;            // one wave32 per batch row
  const float* irow = h2 + (size_t)b * C;
  float acc[10];
  #pragma unroll
  for (int o = 0; o < 10; ++o) acc[o] = 0.f;
  for (int c = lane; c < C; c += 32) {
    float v = (irow[c] - mean[c]) * istd[c] * gamma[c] + beta[c];
    #pragma unroll
    for (int o = 0; o < 10; ++o) acc[o] += v * w3[o * C + c];
  }
  #pragma unroll
  for (int o = 0; o < 10; ++o) {
    float a = acc[o];
    for (int off = 16; off > 0; off >>= 1) a += __shfl_down(a, off);
    if (lane == 0) out[(size_t)b * 10 + o] = a;
  }
}

extern "C" void kernel_launch(void* const* d_in, const int* in_sizes, int n_in,
                              void* d_out, int out_size, void* d_ws, size_t ws_size,
                              hipStream_t stream)
{
  const float* x  = (const float*)d_in[0];
  const float* w1 = (const float*)d_in[1];
  const float* g1 = (const float*)d_in[2];
  const float* b1 = (const float*)d_in[3];
  const float* w2 = (const float*)d_in[4];
  const float* g2 = (const float*)d_in[5];
  const float* b2 = (const float*)d_in[6];
  const float* w3 = (const float*)d_in[7];
  float* out = (float*)d_out;

  const int D = 3072, C = 300, Cpad = 320;
  const int Bn = in_sizes[0] / D;          // 16384

  // workspace sub-allocation (256B aligned)
  char* ws = (char*)d_ws;
  size_t off = 0;
  auto alloc = [&](size_t bytes) -> void* {
    void* p = ws + off;
    off = (off + bytes + 255) & ~(size_t)255;
    return p;
  };
  h16*   xn  = (h16*)  alloc((size_t)Bn * D * sizeof(h16));      // normalized x, f16
  h16*   w1n = (h16*)  alloc((size_t)Cpad * D * sizeof(h16));    // normalized w1^T-ready, padded
  float* h1  = (float*)alloc((size_t)Bn * C * sizeof(float));
  h16*   h1n = (h16*)  alloc((size_t)Bn * Cpad * sizeof(h16));   // BN+normalized h1, f16
  h16*   w2n = (h16*)  alloc((size_t)Cpad * Cpad * sizeof(h16));
  float* h2  = (float*)alloc((size_t)Bn * C * sizeof(float));
  float* mu1 = (float*)alloc(2 * C * sizeof(float)); float* is1 = mu1 + C;
  float* mu2 = (float*)alloc(2 * C * sizeof(float)); float* is2 = mu2 + C;
  (void)ws_size; (void)n_in; (void)out_size;

  // stage operands: unit-norm rows in f16
  norm_rows_f16<<<Bn,   256, 0, stream>>>(x,  xn,  Bn, D, D);
  norm_rows_f16<<<Cpad, 256, 0, stream>>>(w1, w1n, C,  D, D);      // rows 300..319 zeroed
  norm_rows_f16<<<Cpad, 256, 0, stream>>>(w2, w2n, C,  C, Cpad);   // K and N padded

  // fc1: cos-sim + clamp
  wmma_gemm_cos<<<Bn / 32, 320, 0, stream>>>(xn, w1n, h1, D, C, C, 1);
  // bn1 stats, then fused bn1-apply + row-normalize for fc2
  bn_stats<<<C, 256, 0, stream>>>(h1, Bn, C, mu1, is1);
  bn_norm_rows_f16<<<Bn, 128, 0, stream>>>(h1, mu1, is1, g1, b1, h1n, C, Cpad);

  // fc2: cos-sim + clamp
  wmma_gemm_cos<<<Bn / 32, 320, 0, stream>>>(h1n, w2n, h2, Cpad, C, C, 1);
  // bn2 stats, then fused bn2-apply + fc3
  bn_stats<<<C, 256, 0, stream>>>(h2, Bn, C, mu2, is2);
  bn_fc3<<<Bn, 32, 0, stream>>>(h2, mu2, is2, g2, b2, w3, out, C);
}
